// WanSelfAttention_Sparse_40707700032196
// MI455X (gfx1250) — compile-verified, matrix operand order checked by reading, not run on hardware
//
#include <hip/hip_runtime.h>
#include <hip/hip_bf16.h>
#include <stdint.h>

#define S_LEN 4096
#define DIMC  1536
#define NHEADS 12
#define HDIM  128

typedef __attribute__((ext_vector_type(16))) __bf16 v16bf;
typedef __attribute__((ext_vector_type(8)))  float  v8f;
typedef int vs4i __attribute__((vector_size(16)));   // matches builtin prototype

union Frag16 { v16bf v; uint4 u[2]; };

__device__ __forceinline__ unsigned short f2bf(float f) {
  union { float f; unsigned int u; } x; x.f = f;
  unsigned int r = x.u + 0x7FFFu + ((x.u >> 16) & 1u);
  return (unsigned short)(r >> 16);
}
__device__ __forceinline__ float bf2f(unsigned short h) {
  union { unsigned int u; float f; } x; x.u = ((unsigned int)h) << 16;
  return x.f;
}

// ---------- CDNA5 async global->LDS DMA (ASYNCcnt-tracked), with fallback ----------
#if __has_builtin(__builtin_amdgcn_global_load_async_to_lds_b128)
__device__ __forceinline__ void async_copy_b128(const void* g, void* l) {
  __builtin_amdgcn_global_load_async_to_lds_b128(
      (__attribute__((address_space(1))) vs4i*)(unsigned long long)(uintptr_t)g,
      (__attribute__((address_space(3))) vs4i*)(unsigned int)(uintptr_t)l,
      0, 0);
}
#else
__device__ __forceinline__ void async_copy_b128(const void* g, void* l) {
  *(uint4*)l = *(const uint4*)g;   // synchronous fallback
}
#endif

#if __has_builtin(__builtin_amdgcn_s_wait_asynccnt)
#define WAIT_ASYNC() __builtin_amdgcn_s_wait_asynccnt(0)
#else
#define WAIT_ASYNC() asm volatile("s_wait_asynccnt 0x0" ::: "memory")
#endif

// ---------------- fp32 -> bf16 bulk convert ----------------
__global__ __launch_bounds__(256) void cvt_f32_bf16(const float* __restrict__ in,
                                                    unsigned short* __restrict__ out,
                                                    int n) {
  int i = (blockIdx.x * 256 + threadIdx.x) * 4;
  if (i + 3 < n) {
    float4 v = *(const float4*)(in + i);
    out[i + 0] = f2bf(v.x); out[i + 1] = f2bf(v.y);
    out[i + 2] = f2bf(v.z); out[i + 3] = f2bf(v.w);
  }
}

// ---------------- bf16 GEMM: C = A(MxK) * W(NxK)^T + bias ----------------
// Block tile 128x128, BK=32, double-buffered LDS staged by async DMA.
// 256 threads = 8 waves in 4x2 grid; each wave computes 32x64 = 2x4 wmma tiles.
template <bool F32OUT>
__global__ __launch_bounds__(256) void gemm_bf16(const unsigned short* __restrict__ A,
                                                 const unsigned short* __restrict__ W,
                                                 const float* __restrict__ bias,
                                                 unsigned short* __restrict__ Cbf,
                                                 float* __restrict__ Cf,
                                                 int M, int N, int K) {
  __shared__ __align__(16) unsigned short As[2][128 * 32];
  __shared__ __align__(16) unsigned short Bs[2][128 * 32];

  const int t    = threadIdx.x;
  const int wave = t >> 5;
  const int lane = t & 31;
  const int waveM = wave >> 1;          // 0..3
  const int waveN = wave & 1;           // 0..1
  const int bm = blockIdx.x * 128;
  const int bn = blockIdx.y * 128;
  const int lm = lane & 15;
  const int hi = lane >> 4;
  const int kb = hi * 8;

  const int lrow = t >> 1;              // 0..127
  const int lcol = (t & 1) * 16;        // 0 or 16

  const unsigned short* gA = A + (size_t)(bm + lrow) * K + lcol;
  const unsigned short* gW = W + (size_t)(bn + lrow) * K + lcol;

  auto stage = [&](int buf, int k0) {
    async_copy_b128(gA + k0,     &As[buf][lrow * 32 + lcol]);
    async_copy_b128(gA + k0 + 8, &As[buf][lrow * 32 + lcol + 8]);
    async_copy_b128(gW + k0,     &Bs[buf][lrow * 32 + lcol]);
    async_copy_b128(gW + k0 + 8, &Bs[buf][lrow * 32 + lcol + 8]);
  };

  v8f acc[2][4] = {};

  stage(0, 0);
  WAIT_ASYNC();
  __syncthreads();

  for (int k0 = 0; k0 < K; k0 += 32) {
    const int cur = (k0 >> 5) & 1;
    const int nxt = cur ^ 1;
    if (k0 + 32 < K) stage(nxt, k0 + 32);   // DMA next tile while computing

    Frag16 af[2], bfr[4];
#pragma unroll
    for (int r = 0; r < 2; ++r) {
      int row = waveM * 32 + r * 16 + lm;
      af[r].u[0] = *(const uint4*)&As[cur][row * 32 + kb];
      af[r].u[1] = *(const uint4*)&As[cur][row * 32 + 16 + kb];
    }
#pragma unroll
    for (int c = 0; c < 4; ++c) {
      int col = waveN * 64 + c * 16 + lm;
      bfr[c].u[0] = *(const uint4*)&Bs[cur][col * 32 + kb];
      bfr[c].u[1] = *(const uint4*)&Bs[cur][col * 32 + 16 + kb];
    }
#pragma unroll
    for (int r = 0; r < 2; ++r)
#pragma unroll
      for (int c = 0; c < 4; ++c)
        acc[r][c] = __builtin_amdgcn_wmma_f32_16x16x32_bf16(
            false, af[r].v, false, bfr[c].v, (short)0, acc[r][c], false, false);

    if (k0 + 32 < K) WAIT_ASYNC();
    __syncthreads();
  }

#pragma unroll
  for (int r = 0; r < 2; ++r)
#pragma unroll
    for (int c = 0; c < 4; ++c)
#pragma unroll
      for (int i = 0; i < 8; ++i) {
        int row = bm + waveM * 32 + r * 16 + i + hi * 8;
        int col = bn + waveN * 64 + c * 16 + lm;
        float v = acc[r][c][i] + bias[col];
        if constexpr (F32OUT) Cf[(size_t)row * N + col] = v;
        else                  Cbf[(size_t)row * N + col] = f2bf(v);
      }
}

// ---------------- per-token rmsnorm + RoPE on Q/K (in place, bf16) ----------------
__global__ __launch_bounds__(256) void qk_post(unsigned short* __restrict__ qh,
                                               const float* __restrict__ g,
                                               const float* __restrict__ freqs) {
  __shared__ float rowv[DIMC];
  __shared__ float red[8];
  const int s = blockIdx.x;
  const int t = threadIdx.x;

  float ss = 0.f;
  for (int i = t; i < DIMC; i += 256) {
    float v = bf2f(qh[(size_t)s * DIMC + i]);
    rowv[i] = v;
    ss += v * v;
  }
#pragma unroll
  for (int off = 16; off >= 1; off >>= 1) ss += __shfl_xor(ss, off, 32);
  if ((t & 31) == 0) red[t >> 5] = ss;
  __syncthreads();
  float total = 0.f;
#pragma unroll
  for (int i = 0; i < 8; ++i) total += red[i];
  const float rms = rsqrtf(total * (1.0f / DIMC) + 1e-6f);

  // token -> (f,h,w) in 4x32x32 grid; angle split 22/21/21 over 64 pair-dims
  const int fp = s >> 10;
  const int hp = (s >> 5) & 31;
  const int wp = s & 31;
#pragma unroll
  for (int j = 0; j < 3; ++j) {
    int p = t + 256 * j;                 // 0..767 pair index
    int head = p >> 6;
    int i = p & 63;
    int pos = (i < 22) ? fp : ((i < 43) ? hp : wp);
    float ang = freqs[pos * 64 + i];
    float cv = __cosf(ang), sv = __sinf(ang);
    int d0 = head * HDIM + 2 * i;
    float xe = rowv[d0] * rms * g[d0];
    float xo = rowv[d0 + 1] * rms * g[d0 + 1];
    qh[(size_t)s * DIMC + d0]     = f2bf(xe * cv - xo * sv);
    qh[(size_t)s * DIMC + d0 + 1] = f2bf(xe * sv + xo * cv);
  }
}

// ---------------- flash attention: one block per (128 q-rows, head) ----------------
// Double-buffered K/V tiles; K staged by async DMA, V transposed through registers.
__global__ __launch_bounds__(256) void attn_flash(const unsigned short* __restrict__ qh,
                                                  const unsigned short* __restrict__ kh,
                                                  const unsigned short* __restrict__ vh,
                                                  unsigned short* __restrict__ oh) {
  __shared__ __align__(16) unsigned short Ks[2][32 * 128];   // [key][d]
  __shared__ __align__(16) unsigned short Vt[2][128 * 32];   // [d][key] (transposed)
  __shared__ __align__(16) unsigned short Pb[8][16 * 32];    // per-wave P bounce

  const int t    = threadIdx.x;
  const int wave = t >> 5;
  const int lane = t & 31;
  const int lm = lane & 15;
  const int hi = lane >> 4;
  const int kb = hi * 8;
  const int h = blockIdx.y;
  const int qbase = blockIdx.x * 128 + wave * 16;

  // Q resident as 4 bf16 A-fragments covering HD=128
  Frag16 qa[4];
  const unsigned short* qrow = qh + (size_t)(qbase + lm) * DIMC + h * HDIM;
#pragma unroll
  for (int ck = 0; ck < 4; ++ck) {
    qa[ck].u[0] = *(const uint4*)(qrow + ck * 32 + kb);
    qa[ck].u[1] = *(const uint4*)(qrow + ck * 32 + 16 + kb);
  }

  v8f oacc[8] = {};
  float rm[8], rl[8];
#pragma unroll
  for (int i = 0; i < 8; ++i) { rm[i] = -INFINITY; rl[i] = 0.f; }
  const float scale = 0.08838834764831845f;  // 1/sqrt(128)

  const int r  = t >> 3;          // 0..31 (key row of tile)
  const int c0 = (t & 7) * 16;    // d-chunk

  auto stageKV = [&](int buf, int j) {
    const unsigned short* kg = kh + (size_t)(j * 32 + r) * DIMC + h * HDIM + c0;
    async_copy_b128(kg,     &Ks[buf][r * 128 + c0]);
    async_copy_b128(kg + 8, &Ks[buf][r * 128 + c0 + 8]);
    const unsigned short* vg = vh + (size_t)(j * 32 + r) * DIMC + h * HDIM + c0;
    uint4 v0 = ((const uint4*)vg)[0], v1 = ((const uint4*)vg)[1];
    const unsigned short* e = (const unsigned short*)&v0;
#pragma unroll
    for (int q = 0; q < 8; ++q) Vt[buf][(c0 + q) * 32 + r] = e[q];
    e = (const unsigned short*)&v1;
#pragma unroll
    for (int q = 0; q < 8; ++q) Vt[buf][(c0 + 8 + q) * 32 + r] = e[q];
  };

  const int NJ = S_LEN / 32;
  stageKV(0, 0);
  WAIT_ASYNC();
  __syncthreads();

  for (int j = 0; j < NJ; ++j) {
    const int cur = j & 1;
    const int nxt = cur ^ 1;
    if (j + 1 < NJ) stageKV(nxt, j + 1);   // prefetch next tile during compute

    // S = Q * K^T for two 16-key sub-tiles
    v8f sc[2];
#pragma unroll
    for (int tn = 0; tn < 2; ++tn) {
      sc[tn] = (v8f){};
#pragma unroll
      for (int ck = 0; ck < 4; ++ck) {
        Frag16 bfrag;
        int krow = tn * 16 + lm;
        bfrag.u[0] = *(const uint4*)&Ks[cur][krow * 128 + ck * 32 + kb];
        bfrag.u[1] = *(const uint4*)&Ks[cur][krow * 128 + ck * 32 + 16 + kb];
        sc[tn] = __builtin_amdgcn_wmma_f32_16x16x32_bf16(
            false, qa[ck].v, false, bfrag.v, (short)0, sc[tn], false, false);
      }
    }

    // online softmax (row = i + 8*hi across vgpr index, cols across 16 lanes)
    float p0[8], p1[8], corr[8];
#pragma unroll
    for (int i = 0; i < 8; ++i) {
      float s0 = sc[0][i] * scale, s1 = sc[1][i] * scale;
      float mx = fmaxf(s0, s1);
#pragma unroll
      for (int off = 1; off < 16; off <<= 1) mx = fmaxf(mx, __shfl_xor(mx, off, 32));
      float nm = fmaxf(rm[i], mx);
      float cr = __expf(rm[i] - nm);
      rm[i] = nm;
      float e0 = __expf(s0 - nm), e1 = __expf(s1 - nm);
      float rs = e0 + e1;
#pragma unroll
      for (int off = 1; off < 16; off <<= 1) rs += __shfl_xor(rs, off, 32);
      rl[i] = rl[i] * cr + rs;
      corr[i] = cr; p0[i] = e0; p1[i] = e1;
    }
#pragma unroll
    for (int c = 0; c < 8; ++c)
#pragma unroll
      for (int i = 0; i < 8; ++i) oacc[c][i] *= corr[i];

    // relayout P (C-layout) -> A-fragment via per-wave LDS bounce
    unsigned short* pb = Pb[wave];
#pragma unroll
    for (int i = 0; i < 8; ++i) {
      pb[(i + hi * 8) * 32 + lm]      = f2bf(p0[i]);
      pb[(i + hi * 8) * 32 + 16 + lm] = f2bf(p1[i]);
    }
    asm volatile("s_wait_dscnt 0x0" ::: "memory");  // same-wave LDS RAW
    Frag16 pf;
    pf.u[0] = *(const uint4*)&pb[lm * 32 + kb];
    pf.u[1] = *(const uint4*)&pb[lm * 32 + 16 + kb];

    // O += P * V   (K=32, 8 n-chunks over HD=128)
#pragma unroll
    for (int c = 0; c < 8; ++c) {
      Frag16 vf;
      int dcol = c * 16 + lm;
      vf.u[0] = *(const uint4*)&Vt[cur][dcol * 32 + kb];
      vf.u[1] = *(const uint4*)&Vt[cur][dcol * 32 + 16 + kb];
      oacc[c] = __builtin_amdgcn_wmma_f32_16x16x32_bf16(
          false, pf.v, false, vf.v, (short)0, oacc[c], false, false);
    }

    if (j + 1 < NJ) WAIT_ASYNC();
    __syncthreads();
  }

#pragma unroll
  for (int c = 0; c < 8; ++c)
#pragma unroll
    for (int i = 0; i < 8; ++i) {
      int row = qbase + i + hi * 8;
      int col = h * HDIM + c * 16 + lm;
      oh[(size_t)row * DIMC + col] = f2bf(oacc[c][i] / rl[i]);
    }
}

// ---------------- launcher ----------------
extern "C" void kernel_launch(void* const* d_in, const int* in_sizes, int n_in,
                              void* d_out, int out_size, void* d_ws, size_t ws_size,
                              hipStream_t stream) {
  (void)in_sizes; (void)n_in; (void)out_size; (void)ws_size;
  const float* x     = (const float*)d_in[0];
  const float* freqs = (const float*)d_in[3];
  const float* Wq = (const float*)d_in[5];  const float* bq = (const float*)d_in[6];
  const float* Wk = (const float*)d_in[7];  const float* bk = (const float*)d_in[8];
  const float* Wv = (const float*)d_in[9];  const float* bv = (const float*)d_in[10];
  const float* Wo = (const float*)d_in[11]; const float* bo = (const float*)d_in[12];
  const float* gq = (const float*)d_in[13]; const float* gk = (const float*)d_in[14];

  char* ws = (char*)d_ws;
  const size_t SD = (size_t)S_LEN * DIMC * sizeof(unsigned short);  // 12.6 MB
  const size_t WW = (size_t)DIMC * DIMC * sizeof(unsigned short);   //  4.7 MB
  unsigned short* xh  = (unsigned short*)(ws);
  unsigned short* wqh = (unsigned short*)(ws + SD);
  unsigned short* wkh = (unsigned short*)(ws + SD + WW);
  unsigned short* wvh = (unsigned short*)(ws + SD + 2 * WW);
  unsigned short* woh = (unsigned short*)(ws + SD + 3 * WW);
  unsigned short* qhb = (unsigned short*)(ws + SD + 4 * WW);
  unsigned short* khb = (unsigned short*)(ws + 2 * SD + 4 * WW);
  unsigned short* vhb = (unsigned short*)(ws + 3 * SD + 4 * WW);
  unsigned short* ohb = (unsigned short*)(ws + 4 * SD + 4 * WW);

  const int nx = S_LEN * DIMC, nw = DIMC * DIMC;
  cvt_f32_bf16<<<nx / 4 / 256, 256, 0, stream>>>(x, xh, nx);
  cvt_f32_bf16<<<nw / 4 / 256, 256, 0, stream>>>(Wq, wqh, nw);
  cvt_f32_bf16<<<nw / 4 / 256, 256, 0, stream>>>(Wk, wkh, nw);
  cvt_f32_bf16<<<nw / 4 / 256, 256, 0, stream>>>(Wv, wvh, nw);
  cvt_f32_bf16<<<nw / 4 / 256, 256, 0, stream>>>(Wo, woh, nw);

  dim3 gg(S_LEN / 128, DIMC / 128);
  gemm_bf16<false><<<gg, 256, 0, stream>>>(xh, wqh, bq, qhb, nullptr, S_LEN, DIMC, DIMC);
  gemm_bf16<false><<<gg, 256, 0, stream>>>(xh, wkh, bk, khb, nullptr, S_LEN, DIMC, DIMC);
  gemm_bf16<false><<<gg, 256, 0, stream>>>(xh, wvh, bv, vhb, nullptr, S_LEN, DIMC, DIMC);

  qk_post<<<S_LEN, 256, 0, stream>>>(qhb, gq, freqs);
  qk_post<<<S_LEN, 256, 0, stream>>>(khb, gk, freqs);

  attn_flash<<<dim3(S_LEN / 128, NHEADS), 256, 0, stream>>>(qhb, khb, vhb, ohb);

  gemm_bf16<true><<<gg, 256, 0, stream>>>(ohb, woh, bo, nullptr, (float*)d_out,
                                          S_LEN, DIMC, DIMC);
}
